// FunctionCorrelation_40209483825163
// MI455X (gfx1250) — matrix operand order repaired and checked by reading, not compiled, not run on hardware
//
#include <hip/hip_runtime.h>

// CDNA5 (gfx1250) wave32 WMMA types
typedef __attribute__((ext_vector_type(16))) __bf16 v16bf;
typedef __attribute__((ext_vector_type(8)))  float  v8f;

#define Bn 8
#define Cn 256
#define Hn 96
#define Wn 128
#define HWn (Hn * Wn)
#define MDISP 4                    // max displacement; 9x9 = 81 channels
#define NCH 81
#define YB 4                       // y-rows per block
#define NROWS (YB + 2 * MDISP)     // 12 second-rows needed per block
#define FRAG_DW 2048               // dwords per fragment set: 8 chunks * 32 lanes * 8
#define A_DW (YB * FRAG_DW)        // 8192  dwords (32 KB)
#define B_DW (NROWS * 2 * FRAG_DW) // 49152 dwords (192 KB)
#define NTHREADS 288               // 9 waves: one per dy in [-4, 4]

// Pack two f32 into one dword of two bf16 (round-to-nearest-even), low = a.
__device__ __forceinline__ unsigned bf16pack2(float a, float b) {
    unsigned ua = __builtin_bit_cast(unsigned, a);
    unsigned ub = __builtin_bit_cast(unsigned, b);
    ua = ua + 0x7FFFu + ((ua >> 16) & 1u);
    ub = ub + 0x7FFFu + ((ub >> 16) & 1u);
    return (ua >> 16) | (ub & 0xFFFF0000u);
}

// WMMA 16-bit fragment dword mapping (A 16x32 / B 32x16, ISA 7.12.2):
//   lane L: row/col = L&15, half = L>>4
//   dword d (0..7) holds channels c0, c0+1 with
//   c0 = q*32 + ((d>=4)?16:0) + half*8 + ((2d)&7)
__device__ __forceinline__ int frag_ch(int q, int d, int hf) {
    return q * 32 + ((d >= 4) ? 16 : 0) + hf * 8 + ((2 * d) & 7);
}

__global__ __launch_bounds__(NTHREADS) void corr81_wmma_bf16(
        const float* __restrict__ first,
        const float* __restrict__ second,
        float* __restrict__ out) {
    extern __shared__ unsigned smem[];
    unsigned* a_lds = smem;          // [yl][q][lane][d]
    unsigned* b_lds = smem + A_DW;   // [r][t][q][lane][d]

    const int tid = threadIdx.x;
    const int x0  = blockIdx.x * 16;
    const int y0  = (int)blockIdx.y * YB;
    const int b   = (int)blockIdx.z;

    const float* fb = first  + (size_t)b * Cn * HWn;
    const float* sb = second + (size_t)b * Cn * HWn;

    // ---- Stage A fragments: first[b, :, y0:y0+YB, x0:x0+16] (coalesced over low 4 bits) ----
    for (int e = tid; e < A_DW; e += NTHREADS) {
        const int l16 = (e >> 3) & 15;
        const int hf  = (e >> 7) & 1;
        const int d   = e & 7;
        const int q   = (e >> 8) & 7;
        const int yl  = e >> 11;
        const int c0  = frag_ch(q, d, hf);
        const size_t o = (size_t)c0 * HWn + (size_t)(y0 + yl) * Wn + (x0 + l16);
        a_lds[e] = bf16pack2(fb[o], fb[o + HWn]);
    }

    // ---- Stage B fragments: second rows [y0-4, y0+7], x' tiles [x0-4,x0+12) and [x0+12,x0+28) ----
    for (int e = tid; e < B_DW; e += NTHREADS) {
        const int l16 = (e >> 3) & 15;
        const int hf  = (e >> 7) & 1;
        const int d   = e & 7;
        const int q   = (e >> 8) & 7;
        const int rt  = e >> 11;       // 0..23
        const int t   = rt & 1;        // tile
        const int r   = rt >> 1;       // row 0..11
        const int ys  = y0 - MDISP + r;
        const int xp  = x0 + (t ? 12 : -MDISP) + l16;
        const bool ok = (ys >= 0) && (ys < Hn) && (xp >= 0) && (xp < Wn);
        const int ysc = (ys < 0) ? 0 : ((ys > Hn - 1) ? (Hn - 1) : ys);
        const int xpc = (xp < 0) ? 0 : ((xp > Wn - 1) ? (Wn - 1) : xp);
        const int c0  = frag_ch(q, d, hf);
        const size_t o = (size_t)c0 * HWn + (size_t)ysc * Wn + xpc;
        float f0 = sb[o];
        float f1 = sb[o + HWn];
        f0 = ok ? f0 : 0.0f;
        f1 = ok ? f1 : 0.0f;
        b_lds[e] = bf16pack2(f0, f1);
    }
    __syncthreads();

    const int lane = tid & 31;
    const int wid  = tid >> 5;          // 0..8
    const int dy   = wid - MDISP;       // -4..4
    const int half = lane >> 4;
    const int l16  = lane & 15;
    const float scale = 1.0f / (float)Cn;

    for (int yl = 0; yl < YB; ++yl) {
        const int r = yl + wid;         // (y0+yl+dy) - (y0-4), in [0, 11]

        v8f d1 = {};
        v8f d2 = {};
        #pragma unroll
        for (int q = 0; q < 8; ++q) {
            // Pure LDS steady state: 6x ds_load_b128 + 2x WMMA per chunk
            const v16bf av  = *reinterpret_cast<const v16bf*>(
                                  &a_lds[((yl * 8 + q) * 32 + lane) * 8]);
            const v16bf b1v = *reinterpret_cast<const v16bf*>(
                                  &b_lds[(((r * 2 + 0) * 8 + q) * 32 + lane) * 8]);
            const v16bf b2v = *reinterpret_cast<const v16bf*>(
                                  &b_lds[(((r * 2 + 1) * 8 + q) * 32 + lane) * 8]);
            d1 = __builtin_amdgcn_wmma_f32_16x16x32_bf16(false, av, false, b1v,
                                                         (short)0, d1, false, false);
            d2 = __builtin_amdgcn_wmma_f32_16x16x32_bf16(false, av, false, b2v,
                                                         (short)0, d2, false, false);
        }

        // C/D 16x16 f32 layout: lane 0-15 -> N=lane, rows M=0..7 in v0..v7;
        //                       lane 16-31 -> N=lane-16, rows M=8..15.
        float* obase = out + (((size_t)b * NCH) * Hn + (size_t)(y0 + yl)) * Wn;
        #pragma unroll
        for (int v = 0; v < 8; ++v) {
            const int m = half * 8 + v;   // x-local
            const int x = x0 + m;
            const int dx1 = l16 - m - MDISP;           // tile1 band
            if (dx1 >= -MDISP && dx1 <= MDISP) {
                const int ch = (dy + MDISP) * 9 + (dx1 + MDISP);
                obase[(size_t)ch * HWn + x] = d1[v] * scale;
            }
            const int dx2 = l16 + 12 - m;              // tile2 band
            if (dx2 >= -MDISP && dx2 <= MDISP) {
                const int ch = (dy + MDISP) * 9 + (dx2 + MDISP);
                obase[(size_t)ch * HWn + x] = d2[v] * scale;
            }
        }
    }
}

extern "C" void kernel_launch(void* const* d_in, const int* in_sizes, int n_in,
                              void* d_out, int out_size, void* d_ws, size_t ws_size,
                              hipStream_t stream) {
    const float* first  = (const float*)d_in[0];
    const float* second = (const float*)d_in[1];
    float* out = (float*)d_out;
    (void)in_sizes; (void)n_in; (void)out_size; (void)d_ws; (void)ws_size;

    dim3 grid(Wn / 16, Hn / YB, Bn);
    dim3 block(NTHREADS, 1, 1);
    const size_t shmem = (size_t)(A_DW + B_DW) * sizeof(unsigned);   // 224 KB of WGP LDS
    corr81_wmma_bf16<<<grid, block, shmem, stream>>>(first, second, out);
}
